// Decoder_68032281969261
// MI455X (gfx1250) — matrix operand order repaired and checked by reading, not compiled, not run on hardware
//
#include <hip/hip_runtime.h>

// ---------------------------------------------------------------- constants
#define B_   64
#define L_   512
#define E_   256
#define H_   256
#define R_   50
#define K_   768           // conv GEMM depth: 3 taps * 256 enc channels
#define NEG_ 1e10f

typedef _Float16 f16;
typedef __attribute__((ext_vector_type(16))) _Float16 v16h;
typedef __attribute__((ext_vector_type(8)))  float    v8f;
typedef __attribute__((ext_vector_type(4)))  unsigned u32x4;
typedef __attribute__((ext_vector_type(4)))  int      i32x4;
typedef __attribute__((ext_vector_type(8)))  int      i32x8;

#if defined(__has_builtin)
#if __has_builtin(__builtin_amdgcn_tensor_load_to_lds) && __has_builtin(__builtin_amdgcn_s_wait_tensorcnt)
#define HAVE_TDM 1
#endif
#endif
#ifndef HAVE_TDM
#define HAVE_TDM 0
#endif

__device__ __forceinline__ float sigm(float x) { return 1.0f / (1.0f + __expf(-x)); }
__device__ __forceinline__ float gelu_exact(float x) {
    return 0.5f * x * (1.0f + erff(x * 0.70710678118654752440f));
}

// ---------------------------------------------------------------- prep kernels
__global__ void to_f16_kernel(const float* __restrict__ src, f16* __restrict__ dst) {
    const size_t i = (size_t)blockIdx.x * blockDim.x + threadIdx.x;
    dst[i] = (f16)src[i];
}

// pack conv weights (E, 2H, 3) enc-half -> f16 [N=E][K=768], k = tap*256 + cin
__global__ void pack_w_kernel(const float* __restrict__ cw, f16* __restrict__ Bp) {
    const int idx = blockIdx.x * blockDim.x + threadIdx.x;   // E_*K_
    const int o = idx / K_;
    const int k = idx % K_;
    const int tap = k >> 8;
    const int cin = k & 255;
    Bp[idx] = (f16)cw[(size_t)o * 1536 + (size_t)cin * 3 + tap];
}

// ---------------------------------------------------------------- make_x
__global__ void make_x_sos(const float* __restrict__ sos, float* __restrict__ x) {
    const int i = blockIdx.x * blockDim.x + threadIdx.x;     // B_*E_
    x[i] = sos[i & (E_ - 1)];
}
__global__ void make_x_span(const float* __restrict__ enc, const int* __restrict__ K1,
                            const int* __restrict__ K2, float* __restrict__ x) {
    const int i = blockIdx.x * blockDim.x + threadIdx.x;
    const int b = i >> 8, e = i & (E_ - 1);
    x[i] = enc[((size_t)(b * L_ + K1[b])) * E_ + e] + enc[((size_t)(b * L_ + K2[b])) * E_ + e];
}
__global__ void make_x_rel(const float* __restrict__ rel_emb, const int* __restrict__ Rin,
                           float* __restrict__ x) {
    const int i = blockIdx.x * blockDim.x + threadIdx.x;
    const int b = i >> 8;
    x[i] = rel_emb[(size_t)Rin[b] * E_ + (i & (E_ - 1))];
}

// ---------------------------------------------------------------- LSTM
__global__ void lstm_gates_kernel(const float* __restrict__ x, const float* __restrict__ hprev,
                                  const float* __restrict__ W_ih, const float* __restrict__ W_hh,
                                  const float* __restrict__ b_ih, const float* __restrict__ b_hh,
                                  float* __restrict__ gates) {
    const int idx = blockIdx.x * blockDim.x + threadIdx.x;   // B_*4H
    const int b = idx >> 10, j = idx & 1023;
    const float* xr = x + b * E_;
    const float* hr = hprev + b * H_;
    const float* wi = W_ih + (size_t)j * E_;
    const float* wh = W_hh + (size_t)j * H_;
    float s = b_ih[j] + b_hh[j];
    for (int k = 0; k < E_; ++k) s += xr[k] * wi[k];
    for (int k = 0; k < H_; ++k) s += hr[k] * wh[k];
    gates[idx] = s;
}

__global__ void lstm_update_kernel(const float* __restrict__ gates, const float* __restrict__ cprev,
                                   float* __restrict__ h, float* __restrict__ c) {
    const int idx = blockIdx.x * blockDim.x + threadIdx.x;   // B_*H
    const int b = idx >> 8, j = idx & 255;
    const float* g = gates + (size_t)b * 4 * H_;
    const float ig = sigm(g[j]);
    const float fg = sigm(g[H_ + j]);
    const float gg = tanhf(g[2 * H_ + j]);
    const float og = sigm(g[3 * H_ + j]);
    const float cn = fg * cprev[idx] + ig * gg;
    c[idx] = cn;
    h[idx] = og * tanhf(cn);
}

// ---------------------------------------------------------------- attention (one block per b)
__global__ void attention_kernel(const float* __restrict__ h, const float* __restrict__ enc,
                                 const float* __restrict__ W_attn, const float* __restrict__ b_attn,
                                 float* __restrict__ outv) {
    __shared__ float hv[E_];
    __shared__ float sc[L_];
    __shared__ float mixs[E_];
    __shared__ float red[256];
    const int b = blockIdx.x, t = threadIdx.x;

    hv[t] = h[b * H_ + t];
    __syncthreads();

    for (int l = t; l < L_; l += 256) {
        const float* e = enc + ((size_t)(b * L_ + l)) * E_;
        float s = 0.0f;
        for (int k = 0; k < E_; ++k) s += hv[k] * e[k];
        sc[l] = s;
    }
    __syncthreads();

    const float v0 = sc[t], v1 = sc[t + 256];
    red[t] = fmaxf(v0, v1);
    __syncthreads();
    for (int s = 128; s > 0; s >>= 1) { if (t < s) red[t] = fmaxf(red[t], red[t + s]); __syncthreads(); }
    const float mx = red[0];
    __syncthreads();
    const float e0 = __expf(v0 - mx), e1 = __expf(v1 - mx);
    red[t] = e0 + e1;
    __syncthreads();
    for (int s = 128; s > 0; s >>= 1) { if (t < s) red[t] += red[t + s]; __syncthreads(); }
    const float inv = 1.0f / red[0];
    __syncthreads();
    sc[t] = e0 * inv;
    sc[t + 256] = e1 * inv;
    __syncthreads();

    // mix[e], e = t
    float acc = 0.0f;
    for (int l = 0; l < L_; ++l) acc += sc[l] * enc[((size_t)(b * L_ + l)) * E_ + t];
    mixs[t] = acc;
    __syncthreads();

    // out[t] = tanh([mix, h] @ W_attn[t,:] + b)
    float s2 = b_attn[t];
    const float* wr = W_attn + (size_t)t * (2 * E_);
    for (int k = 0; k < E_; ++k) s2 += mixs[k] * wr[k];
    for (int k = 0; k < E_; ++k) s2 += hv[k] * wr[E_ + k];
    outv[b * E_ + t] = tanhf(s2);
}

// ---------------------------------------------------------------- out-vector GEMV (conv split)
__global__ void vout_kernel(const float* __restrict__ cw, const float* __restrict__ outv,
                            float* __restrict__ vall, float* __restrict__ vl0,
                            float* __restrict__ vlast) {
    const int idx = blockIdx.x * blockDim.x + threadIdx.x;   // B_*E_
    const int b = idx >> 8, o = idx & 255;
    const float* ov = outv + b * E_;
    const float* w = cw + (size_t)o * 1536 + 256 * 3;        // cin = 256.. half
    float s0 = 0.0f, s1 = 0.0f, s2 = 0.0f;
    for (int c = 0; c < E_; ++c) {
        const float v = ov[c];
        const float* wp = w + c * 3;
        s0 += v * wp[0]; s1 += v * wp[1]; s2 += v * wp[2];
    }
    vall[idx]  = s0 + s1 + s2;
    vl0[idx]   = s1 + s2;        // l==0: tap 0 reads zero padding
    vlast[idx] = s0 + s1;        // l==L-1: tap 2 reads zero padding
}

// ---------------------------------------------------------------- conv as WMMA GEMM
// Block: 256 threads = 8 waves. Block tile = 128 M rows (8 M-tiles, one per wave)
// x 32 N cols (2 N-tiles per wave). The 32x768 f16 B slab (48KB, contiguous in the
// packed layout) is staged into LDS once via the Tensor Data Mover, then each wave
// runs 24 K-chunks of V_WMMA_F32_16X16X32_F16 with the A fragment reused across
// both N-tiles.
__global__ void __launch_bounds__(256)
conv_wmma_kernel(const f16* __restrict__ Af16,      // (B,L,E) f16 enc
                 const f16* __restrict__ Bpack,     // (E,K_) f16 packed weights
                 const float* __restrict__ vall, const float* __restrict__ vl0,
                 const float* __restrict__ vlast, const float* __restrict__ bias,
                 float* __restrict__ enc_out, f16* __restrict__ enc_out_f16,
                 float* __restrict__ a_out) {
    __shared__ f16 smemB[32 * K_];                  // 48 KB

    const int t      = threadIdx.x;
    const int wv     = t >> 5;
    const int lane   = t & 31;
    const int nblock = blockIdx.x & 7;              // 8 N-blocks of 32 cols
    const int mblock = blockIdx.x >> 3;             // 256 M-blocks of 128 rows
    const int n0     = nblock * 32;

    // ---- stage B slab: Bpack rows n0..n0+31 are one contiguous 48KB region
    const f16* gB = Bpack + (size_t)n0 * K_;
#if HAVE_TDM
    if (wv == 0) {
        const unsigned long long ga = (unsigned long long)(uintptr_t)gB;
        const unsigned lds = (unsigned)(uintptr_t)smemB;   // LDS generic addr low bits = byte offset
        const int elems = 32 * K_;                         // 24576 f16 elements (1-D tile)
        u32x4 g0;
        g0[0] = 1u;                                        // count=1, user mode
        g0[1] = lds;                                       // lds_addr (bytes)
        g0[2] = (unsigned)(ga & 0xFFFFFFFFu);              // global_addr[31:0]
        g0[3] = (unsigned)((ga >> 32) & 0x01FFFFFFu) | 0x80000000u; // addr[56:32] | type=2
        i32x8 g1;
        g1[0] = 0x00010000;                                // workgroup_mask=0, data_size=1 (2B)
        g1[1] = (elems & 0xFFFF) << 16;                    // tensor_dim0[15:0]
        g1[2] = ((elems >> 16) & 0xFFFF) | (1 << 16);      // tensor_dim0[31:16] | tensor_dim1=1
        g1[3] = (elems & 0xFFFF) << 16;                    // tile_dim0 = 24576
        g1[4] = 0;                                         // tile_dim1=0 (unused), tile_dim2=0
        g1[5] = elems;                                     // tensor_dim0_stride[31:0]
        g1[6] = 0;
        g1[7] = 0;
        i32x4 z4 = {0, 0, 0, 0};
#if __clang_major__ >= 23
        i32x8 z8 = {0, 0, 0, 0, 0, 0, 0, 0};
        __builtin_amdgcn_tensor_load_to_lds(g0, g1, z4, z4, z8, 0);
#else
        __builtin_amdgcn_tensor_load_to_lds(g0, g1, z4, z4, 0);
#endif
        __builtin_amdgcn_s_wait_tensorcnt(0);
    }
    __syncthreads();
#else
    {
        const float4* src = (const float4*)gB;
        float4* dst = (float4*)smemB;
        for (int i = t; i < (32 * K_ * 2) / 16; i += 256) dst[i] = src[i];
        __syncthreads();
    }
#endif

    // ---- per-wave tile coordinates
    const int mtile = mblock * 8 + wv;              // 0..2047
    const int b     = mtile >> 5;                   // 32 M-tiles per batch row (L/16)
    const int l0    = (mtile & 31) << 4;
    const bool hi   = lane >= 16;
    const int  l    = l0 + (lane & 15);             // A row for this lane

    const int kAoff = hi ? 16 : 0;
    const int kBoff = hi ? 16 : 0;
    const f16* BrowL0 = smemB + (size_t)(lane & 15) * K_ + kBoff;          // N-tile 0
    const f16* BrowL1 = BrowL0 + (size_t)16 * K_;                          // N-tile 1

    __builtin_prefetch(Af16 + ((size_t)(b * L_ + l0)) * E_, 0, 3);

    v8f acc0 = {}, acc1 = {};
    for (int kc = 0; kc < K_ / 32; ++kc) {
        const int k0   = kc * 32;
        const int tap  = k0 >> 8;                   // uniform within a 32-wide K chunk
        const int cin0 = k0 & 255;
        const int lsrc = l + tap - 1;

        v16h afrag = {};
        if (lsrc >= 0 && lsrc < L_) {
            const f16* Ap = Af16 + ((size_t)(b * L_ + lsrc)) * E_ + cin0 + kAoff;
            #pragma unroll
            for (int r = 0; r < 8; ++r) {
                const int off = 2 * r;
                afrag[2 * r]     = Ap[off];
                afrag[2 * r + 1] = Ap[off + 1];
            }
        }
        v16h b0, b1;
        const f16* Bp0 = BrowL0 + k0;
        const f16* Bp1 = BrowL1 + k0;
        #pragma unroll
        for (int r = 0; r < 8; ++r) {
            b0[2 * r] = Bp0[2 * r]; b0[2 * r + 1] = Bp0[2 * r + 1];
            b1[2 * r] = Bp1[2 * r]; b1[2 * r + 1] = Bp1[2 * r + 1];
        }
        acc0 = __builtin_amdgcn_wmma_f32_16x16x32_f16(false, afrag, false, b0,
                                                      (short)0, acc0, false, false);
        acc1 = __builtin_amdgcn_wmma_f32_16x16x32_f16(false, afrag, false, b1,
                                                      (short)0, acc1, false, false);
    }

    // ---- epilogue: C layout VGPR r -> (M = r + (hi?8:0), N = lane&15)
    const int col = lane & 15;
    #pragma unroll
    for (int r = 0; r < 8; ++r) {
        const int row = r + (hi ? 8 : 0);
        const int le  = l0 + row;
        const float* v = (le == 0) ? vl0 : ((le == L_ - 1) ? vlast : vall);
        const size_t ridx = ((size_t)(b * L_ + le)) * E_;
        {
            const int o = n0 + col;
            const float y = acc0[r] + v[b * E_ + o] + bias[o];
            enc_out[ridx + o]     = y;
            enc_out_f16[ridx + o] = (f16)y;
            a_out[ridx + o]       = gelu_exact(y);
        }
        {
            const int o = n0 + 16 + col;
            const float y = acc1[r] + v[b * E_ + o] + bias[o];
            enc_out[ridx + o]     = y;
            enc_out_f16[ridx + o] = (f16)y;
            a_out[ridx + o]       = gelu_exact(y);
        }
    }
}

// ---------------------------------------------------------------- heads
__global__ void heads_kernel(const float* __restrict__ a, const float* __restrict__ w1,
                             const float* __restrict__ b1, const float* __restrict__ w2,
                             const float* __restrict__ b2, float* __restrict__ o1,
                             float* __restrict__ o2) {
    __shared__ float s1[E_], s2[E_];
    const int m = blockIdx.x * blockDim.x + threadIdx.x;   // 0..B*L-1
    s1[threadIdx.x] = w1[threadIdx.x];
    s2[threadIdx.x] = w2[threadIdx.x];
    __syncthreads();
    const float* ap = a + (size_t)m * E_;
    float r1 = b1[0], r2 = b2[0];
    for (int k = 0; k < E_; ++k) { const float v = ap[k]; r1 += v * s1[k]; r2 += v * s2[k]; }
    o1[m] = r1;
    o2[m] = r2;
}

// ---------------------------------------------------------------- relation logits + masked max-pool
__global__ void rel_maxpool_kernel(const float* __restrict__ a, const float* __restrict__ W_rel,
                                   const float* __restrict__ b_rel, const int* __restrict__ T,
                                   float* __restrict__ out) {
    __shared__ float wr[E_];
    __shared__ float red[128];
    const int b = blockIdx.x, r = blockIdx.y, t = threadIdx.x;   // 128 threads
    wr[t]       = W_rel[(size_t)r * E_ + t];
    wr[t + 128] = W_rel[(size_t)r * E_ + t + 128];
    __syncthreads();
    const float br = b_rel[r];
    float mx = -3.4e38f;
    for (int l = t; l < L_; l += 128) {
        const float* ap = a + ((size_t)(b * L_ + l)) * E_;
        float s = br;
        for (int k = 0; k < E_; ++k) s += ap[k] * wr[k];
        if (T[b * L_ + l] <= 0) s -= NEG_;
        mx = fmaxf(mx, s);
    }
    red[t] = mx;
    __syncthreads();
    for (int s = 64; s > 0; s >>= 1) { if (t < s) red[t] = fmaxf(red[t], red[t + s]); __syncthreads(); }
    if (t == 0) out[b * R_ + r] = red[0];
}

// ---------------------------------------------------------------- host launcher
extern "C" void kernel_launch(void* const* d_in, const int* in_sizes, int n_in,
                              void* d_out, int out_size, void* d_ws, size_t ws_size,
                              hipStream_t stream) {
    (void)in_sizes; (void)n_in; (void)out_size; (void)ws_size;

    const float* encoder_o  = (const float*)d_in[0];
    const float* h0         = (const float*)d_in[1];
    const float* c0         = (const float*)d_in[2];
    const float* sos        = (const float*)d_in[3];
    const float* rel_emb    = (const float*)d_in[4];
    const float* W_ih       = (const float*)d_in[5];
    const float* W_hh       = (const float*)d_in[6];
    const float* b_ih       = (const float*)d_in[7];
    const float* b_hh       = (const float*)d_in[8];
    const float* W_attn     = (const float*)d_in[9];
    const float* b_attn     = (const float*)d_in[10];
    const float* conv_rel_w = (const float*)d_in[11];
    const float* conv_rel_b = (const float*)d_in[12];
    const float* conv_ent_w = (const float*)d_in[13];
    const float* conv_ent_b = (const float*)d_in[14];
    const float* W_rel      = (const float*)d_in[15];
    const float* b_rel      = (const float*)d_in[16];
    const float* w_e1       = (const float*)d_in[17];
    const float* b_e1       = (const float*)d_in[18];
    const float* w_e2       = (const float*)d_in[19];
    const float* b_e2       = (const float*)d_in[20];
    const int*   T          = (const int*)d_in[21];
    const int*   R_in       = (const int*)d_in[22];
    const int*   S_K1       = (const int*)d_in[23];
    const int*   S_K2       = (const int*)d_in[24];
    float* out = (float*)d_out;

    // workspace carve-up (aligned to 256B)
    char* base = (char*)d_ws;
    size_t off = 0;
    auto take = [&](size_t bytes) -> char* {
        char* p = base + off;
        off = (off + bytes + 255) & ~(size_t)255;
        return p;
    };
    const size_t BLE = (size_t)B_ * L_ * E_;
    float* encA  = (float*)take(BLE * 4);
    float* encB  = (float*)take(BLE * 4);
    f16*   hfA   = (f16*)take(BLE * 2);
    f16*   hfB   = (f16*)take(BLE * 2);
    float* a_buf = (float*)take(BLE * 4);
    f16*   BpEnt = (f16*)take((size_t)E_ * K_ * 2);
    f16*   BpRel = (f16*)take((size_t)E_ * K_ * 2);
    float* xbuf  = (float*)take((size_t)B_ * E_ * 4);
    float* gates = (float*)take((size_t)B_ * 4 * H_ * 4);
    float* hbuf  = (float*)take((size_t)B_ * H_ * 4);
    float* cbuf  = (float*)take((size_t)B_ * H_ * 4);
    float* attn  = (float*)take((size_t)B_ * E_ * 4);
    float* vall  = (float*)take((size_t)B_ * E_ * 4);
    float* vl0   = (float*)take((size_t)B_ * E_ * 4);
    float* vlast = (float*)take((size_t)B_ * E_ * 4);

    const int BL = B_ * L_;
    const dim3 relGrid(B_, R_);

    // per-launch prep (deterministic)
    to_f16_kernel<<<(unsigned)(BLE / 256), 256, 0, stream>>>(encoder_o, hfA);
    pack_w_kernel<<<(E_ * K_) / 256, 256, 0, stream>>>(conv_ent_w, BpEnt);
    pack_w_kernel<<<(E_ * K_) / 256, 256, 0, stream>>>(conv_rel_w, BpRel);

    // ---- step 1: sos -> entity heads
    make_x_sos<<<64, 256, 0, stream>>>(sos, xbuf);
    lstm_gates_kernel<<<256, 256, 0, stream>>>(xbuf, h0, W_ih, W_hh, b_ih, b_hh, gates);
    lstm_update_kernel<<<64, 256, 0, stream>>>(gates, c0, hbuf, cbuf);
    attention_kernel<<<64, 256, 0, stream>>>(hbuf, encoder_o, W_attn, b_attn, attn);
    vout_kernel<<<64, 256, 0, stream>>>(conv_ent_w, attn, vall, vl0, vlast);
    conv_wmma_kernel<<<2048, 256, 0, stream>>>(hfA, BpEnt, vall, vl0, vlast, conv_ent_b,
                                               encA, hfB, a_buf);
    heads_kernel<<<128, 256, 0, stream>>>(a_buf, w_e1, b_e1, w_e2, b_e2, out, out + BL);

    // ---- step 2: entity span -> relation logits (masked max-pool)
    make_x_span<<<64, 256, 0, stream>>>(encA, S_K1, S_K2, xbuf);
    lstm_gates_kernel<<<256, 256, 0, stream>>>(xbuf, hbuf, W_ih, W_hh, b_ih, b_hh, gates);
    lstm_update_kernel<<<64, 256, 0, stream>>>(gates, cbuf, hbuf, cbuf);
    attention_kernel<<<64, 256, 0, stream>>>(hbuf, encA, W_attn, b_attn, attn);
    vout_kernel<<<64, 256, 0, stream>>>(conv_rel_w, attn, vall, vl0, vlast);
    conv_wmma_kernel<<<2048, 256, 0, stream>>>(hfB, BpRel, vall, vl0, vlast, conv_rel_b,
                                               encB, hfA, a_buf);
    rel_maxpool_kernel<<<relGrid, 128, 0, stream>>>(a_buf, W_rel, b_rel, T, out + 2 * BL);

    // ---- step 3: relation embedding -> entity heads
    make_x_rel<<<64, 256, 0, stream>>>(rel_emb, R_in, xbuf);
    lstm_gates_kernel<<<256, 256, 0, stream>>>(xbuf, hbuf, W_ih, W_hh, b_ih, b_hh, gates);
    lstm_update_kernel<<<64, 256, 0, stream>>>(gates, cbuf, hbuf, cbuf);
    attention_kernel<<<64, 256, 0, stream>>>(hbuf, encB, W_attn, b_attn, attn);
    vout_kernel<<<64, 256, 0, stream>>>(conv_ent_w, attn, vall, vl0, vlast);
    conv_wmma_kernel<<<2048, 256, 0, stream>>>(hfA, BpEnt, vall, vl0, vlast, conv_ent_b,
                                               encA, hfB, a_buf);
    heads_kernel<<<128, 256, 0, stream>>>(a_buf, w_e1, b_e1, w_e2, b_e2,
                                          out + 2 * BL + B_ * R_, out + 3 * BL + B_ * R_);
}
